// SlotAttention_16286515986499
// MI455X (gfx1250) — matrix operand order repaired and checked by reading, not compiled, not run on hardware
//
#include <hip/hip_runtime.h>
#include <math.h>

// ---------------------------------------------------------------------------
// Slot Attention for MI455X (gfx1250), wave32 + WMMA f32_16x16x32_f16.
//   B=128, N=1024, D_in=192, H=128, K=3 slots, 3 iterations.
// Heavy GEMMs (input proj, k/v proj, q.k^T, attn.v) run on v_wmma with f16
// operands / f32 accumulation; tiny 3-row GRU/MLP math stays f32 VALU.
// k/v are cached in workspace as f16 (67MB -> resident in the 192MB L2
// across the 3 iterations).
// ---------------------------------------------------------------------------

typedef __attribute__((ext_vector_type(16))) _Float16 v16h;
typedef __attribute__((ext_vector_type(8)))  float    v8f;

#define H_    128
#define DIN_  192
#define NTOK_ 1024
#define B_    128
#define KS_   3

// 16-bit A fragment (16x32 MxK): lane m=l%16, half=l/16, element i -> K
__device__ __forceinline__ int amap(int i, int half) {
  return i + 8 * half + ((i >= 8) ? 8 : 0);
}
// 16-bit B fragment (32x16 KxN): lane n=l%16, half=l/16, element i -> K
__device__ __forceinline__ int bmap(int i, int half) {
  return i + 8 * half + ((i >= 8) ? 8 : 0);
}

__device__ __forceinline__ v8f wmma16(v16h a, v16h b, v8f c) {
  return __builtin_amdgcn_wmma_f32_16x16x32_f16(
      /*neg_a=*/false, a, /*neg_b=*/false, b,
      /*c_mod=*/(short)0, c, /*reuse_a=*/false, /*reuse_b=*/false);
}

// ---------------------------------------------------------------------------
// Kernel 0: convert WMMA-side weights to f16 once.
// ---------------------------------------------------------------------------
__global__ void cvt_w(const float* __restrict__ Wp, const float* __restrict__ Wk,
                      const float* __restrict__ Wv, const float* __restrict__ Wq,
                      _Float16* __restrict__ wp16, _Float16* __restrict__ wk16,
                      _Float16* __restrict__ wv16, _Float16* __restrict__ wq16) {
  int i = blockIdx.x * 256 + threadIdx.x;
  if (i < H_ * DIN_) wp16[i] = (_Float16)Wp[i];
  if (i < H_ * H_) {
    wk16[i] = (_Float16)Wk[i];
    wv16[i] = (_Float16)Wv[i];
    wq16[i] = (_Float16)Wq[i];
  }
}

// ---------------------------------------------------------------------------
// Kernel 1: inputs = LN(X @ Wp^T + bp); k = inputs@Wk^T+bk; v = inputs@Wv^T+bv
// 256 threads = 8 waves; each wave owns a 16-row tile (block = 128 rows).
// Per-wave private LDS slab; same-wave DS ops are in-order -> no barriers.
// ---------------------------------------------------------------------------
__launch_bounds__(256)
__global__ void proj_ln_kv(
    const float* __restrict__ X,                                   // (B*N, 192)
    const _Float16* __restrict__ wp16, const float* __restrict__ bp,
    const float* __restrict__ g_in, const float* __restrict__ b_in,
    const _Float16* __restrict__ wk16, const float* __restrict__ bk,
    const _Float16* __restrict__ wv16, const float* __restrict__ bv,
    _Float16* __restrict__ k16, _Float16* __restrict__ v16) {
  __shared__ float lds[8][16][H_];  // 64 KB: 8 waves x (16 x 128) f32

  const int lane = threadIdx.x & 31;
  const int wave = threadIdx.x >> 5;
  const int m    = lane & 15;
  const int half = lane >> 4;
  const long r0  = (long)blockIdx.x * 128 + wave * 16;

  const v8f z = {};

  // ---- Phase 1: P = X @ Wp^T + bp   (Kdim = 192 = 6 chunks of 32)
  v8f acc[8];
#pragma unroll
  for (int t = 0; t < 8; ++t) acc[t] = z;

  for (int c = 0; c < 6; ++c) {
    v16h a;
    const float* xr = X + (r0 + m) * DIN_ + c * 32;
#pragma unroll
    for (int i = 0; i < 16; ++i) a[i] = (_Float16)xr[amap(i, half)];
#pragma unroll
    for (int t = 0; t < 8; ++t) {
      v16h bb;
      const _Float16* wr = wp16 + (t * 16 + m) * DIN_ + c * 32;
#pragma unroll
      for (int i = 0; i < 16; ++i) bb[i] = wr[bmap(i, half)];
      acc[t] = wmma16(a, bb, acc[t]);
    }
  }
#pragma unroll
  for (int t = 0; t < 8; ++t) {
    int col = t * 16 + m;
    float bpc = bp[col];
#pragma unroll
    for (int j = 0; j < 8; ++j)
      lds[wave][j + 8 * half][col] = acc[t][j] + bpc;  // row M = j + 8*half
  }

  // ---- Phase 2: LayerNorm per row (lane owns row m, cols half*64..+63)
  {
    float s1 = 0.f, s2 = 0.f;
#pragma unroll 8
    for (int cc = 0; cc < 64; ++cc) {
      float x = lds[wave][m][half * 64 + cc];
      s1 += x; s2 += x * x;
    }
    s1 += __shfl_xor(s1, 16, 32);
    s2 += __shfl_xor(s2, 16, 32);
    float mu  = s1 * (1.f / 128.f);
    float var = s2 * (1.f / 128.f) - mu * mu;
    float rs  = rsqrtf(var + 1e-5f);
#pragma unroll 8
    for (int cc = 0; cc < 64; ++cc) {
      int col = half * 64 + cc;
      float x = lds[wave][m][col];
      lds[wave][m][col] = (x - mu) * rs * g_in[col] + b_in[col];
    }
  }

  // ---- Phase 3: k = LN @ Wk^T + bk ; v = LN @ Wv^T + bv (Kdim = 128)
  v8f ka[8], va[8];
#pragma unroll
  for (int t = 0; t < 8; ++t) { ka[t] = z; va[t] = z; }

  for (int c = 0; c < 4; ++c) {
    v16h a;
#pragma unroll
    for (int i = 0; i < 16; ++i)
      a[i] = (_Float16)lds[wave][m][c * 32 + amap(i, half)];
#pragma unroll
    for (int t = 0; t < 8; ++t) {
      v16h bkf, bvf;
      const _Float16* wkr = wk16 + (t * 16 + m) * H_ + c * 32;
      const _Float16* wvr = wv16 + (t * 16 + m) * H_ + c * 32;
#pragma unroll
      for (int i = 0; i < 16; ++i) {
        int kk = bmap(i, half);
        bkf[i] = wkr[kk];
        bvf[i] = wvr[kk];
      }
      ka[t] = wmma16(a, bkf, ka[t]);
      va[t] = wmma16(a, bvf, va[t]);
    }
  }
#pragma unroll
  for (int t = 0; t < 8; ++t) {
    int col = t * 16 + m;
    float bkc = bk[col], bvc = bv[col];
#pragma unroll
    for (int j = 0; j < 8; ++j) {
      long row = r0 + j + 8 * half;
      k16[row * H_ + col] = (_Float16)(ka[t][j] + bkc);
      v16[row * H_ + col] = (_Float16)(va[t][j] + bvc);
    }
  }
}

// ---------------------------------------------------------------------------
// Kernel 2: full 3-iteration slot loop, one block (8 waves) per batch b.
// WMMA for q = s@Wq^T, attn = q@k^T (N=1024) and updates = attn@v.
// GRU + MLP (3x128 rows) in f32 VALU.
// ---------------------------------------------------------------------------
__launch_bounds__(256)
__global__ void slot_iter(
    const float* __restrict__ noise, const float* __restrict__ slot_mu,
    const float* __restrict__ slot_ls,
    const _Float16* __restrict__ wq16, const float* __restrict__ bq,
    const _Float16* __restrict__ k16, const _Float16* __restrict__ v16,
    const float* __restrict__ Wih, const float* __restrict__ bih,
    const float* __restrict__ Whh, const float* __restrict__ bhh,
    const float* __restrict__ g_s, const float* __restrict__ b_s,
    const float* __restrict__ W1, const float* __restrict__ b1,
    const float* __restrict__ W2, const float* __restrict__ b2,
    const float* __restrict__ g_m, const float* __restrict__ b_m,
    float* __restrict__ slots_out, float* __restrict__ attn_out) {
  __shared__ _Float16 q16s[16][H_];        // q padded to 16 rows (rows>=3 zero)
  __shared__ _Float16 attnw[KS_][NTOK_];   // softmaxed attention, f16
  __shared__ float slots_s[KS_][H_];
  __shared__ float s_ln[KS_][H_];
  __shared__ float upd[KS_][H_];
  __shared__ float gis[KS_ * 3 * H_];
  __shared__ float ghs[KS_ * 3 * H_];
  __shared__ float hid[KS_ * 2 * H_];

  const int tid  = threadIdx.x;
  const int lane = tid & 31;
  const int wave = tid >> 5;
  const int m    = lane & 15;
  const int half = lane >> 4;
  const int b    = blockIdx.x;

  // slots = slot_mu + exp(slot_log_sigma) * noise
  for (int o = tid; o < KS_ * H_; o += 256)
    slots_s[o / H_][o % H_] =
        slot_mu[o] + expf(slot_ls[o]) * noise[(long)b * KS_ * H_ + o];
  __syncthreads();

  const float scale = 0.0883883476483184405f;  // 1/sqrt(128)

  for (int it = 0; it < 3; ++it) {
    // ---- s = LN(slots, g_s, b_s): waves 0..2 each own a slot row
    if (wave < KS_) {
      float s1 = 0.f, s2 = 0.f;
#pragma unroll
      for (int p = 0; p < 4; ++p) {
        float x = slots_s[wave][lane + 32 * p];
        s1 += x; s2 += x * x;
      }
#pragma unroll
      for (int d = 1; d < 32; d <<= 1) {
        s1 += __shfl_xor(s1, d, 32);
        s2 += __shfl_xor(s2, d, 32);
      }
      float mu  = s1 * (1.f / 128.f);
      float var = s2 * (1.f / 128.f) - mu * mu;
      float rs  = rsqrtf(var + 1e-5f);
#pragma unroll
      for (int p = 0; p < 4; ++p) {
        int c = lane + 32 * p;
        s_ln[wave][c] = (slots_s[wave][c] - mu) * rs * g_s[c] + b_s[c];
      }
    }
    __syncthreads();

    // ---- q = s @ Wq^T + bq : wave w owns output cols [16w, 16w+16)
    {
      v8f acc = {};
#pragma unroll
      for (int c = 0; c < 4; ++c) {
        v16h a, bb;
#pragma unroll
        for (int i = 0; i < 16; ++i) {
          int d = c * 32 + amap(i, half);
          float sv = s_ln[(m < KS_) ? m : 0][d];
          a[i]  = (m < KS_) ? (_Float16)sv : (_Float16)0.f;   // zero-pad rows >= 3
          bb[i] = wq16[(wave * 16 + m) * H_ + c * 32 + bmap(i, half)];
        }
        acc = wmma16(a, bb, acc);
      }
      int col = wave * 16 + m;
      float bqc = bq[col];
#pragma unroll
      for (int j = 0; j < 8; ++j) {
        int M = j + 8 * half;
        q16s[M][col] = (M < KS_) ? (_Float16)(acc[j] + bqc) : (_Float16)0.f;
      }
    }
    __syncthreads();

    // ---- attn = softmax_over_slots(q @ k^T * scale); wave w owns 8 N-tiles
    {
      v16h qa[4];
#pragma unroll
      for (int c = 0; c < 4; ++c)
#pragma unroll
        for (int i = 0; i < 16; ++i)
          qa[c][i] = q16s[m][c * 32 + amap(i, half)];

#pragma unroll 1
      for (int tt = 0; tt < 8; ++tt) {
        int nt = wave * 8 + tt;
        const _Float16* kb = k16 + ((long)b * NTOK_ + nt * 16 + m) * H_;
        v8f acc = {};
#pragma unroll
        for (int c = 0; c < 4; ++c) {
          v16h bb;
#pragma unroll
          for (int i = 0; i < 16; ++i) bb[i] = kb[c * 32 + bmap(i, half)];
          acc = wmma16(qa[c], bb, acc);
        }
        if (half == 0) {  // rows 0..2 live in VGPRs 0..2 of lanes 0..15
          float a0 = acc[0] * scale, a1 = acc[1] * scale, a2 = acc[2] * scale;
          float mx = fmaxf(a0, fmaxf(a1, a2));
          float e0 = expf(a0 - mx), e1 = expf(a1 - mx), e2 = expf(a2 - mx);
          float inv = 1.f / (e0 + e1 + e2);
          e0 *= inv; e1 *= inv; e2 *= inv;
          int nc = nt * 16 + m;
          attn_out[((long)b * KS_ + 0) * NTOK_ + nc] = e0;
          attn_out[((long)b * KS_ + 1) * NTOK_ + nc] = e1;
          attn_out[((long)b * KS_ + 2) * NTOK_ + nc] = e2;
          attnw[0][nc] = (_Float16)e0;
          attnw[1][nc] = (_Float16)e1;
          attnw[2][nc] = (_Float16)e2;
        }
      }
    }
    __syncthreads();

    // ---- updates = attn @ v : wave w owns output cols [16w, 16w+16)
    {
      v8f acc = {};
#pragma unroll 2
      for (int c = 0; c < 32; ++c) {
        v16h a, bb;
#pragma unroll
        for (int i = 0; i < 16; ++i) {
          int d = c * 32 + amap(i, half);
          _Float16 av = attnw[(m < KS_) ? m : 0][d];
          a[i]  = (m < KS_) ? av : (_Float16)0.f;
          bb[i] = v16[((long)b * NTOK_ + c * 32 + bmap(i, half)) * H_ +
                      wave * 16 + m];
        }
        acc = wmma16(a, bb, acc);
      }
#pragma unroll
      for (int j = 0; j < 8; ++j) {
        int M = j + 8 * half;
        if (M < KS_) upd[M][wave * 16 + m] = acc[j];
      }
    }
    __syncthreads();

    // ---- GRU gates: gi = upd@Wih^T+bih ; gh = slots_prev@Whh^T+bhh
    for (int o = tid; o < KS_ * 3 * H_; o += 256) {
      int r = o / (3 * H_), cc = o % (3 * H_);
      const float* wi = Wih + cc * H_;
      const float* wh = Whh + cc * H_;
      float si = bih[cc], sh = bhh[cc];
#pragma unroll 4
      for (int d = 0; d < H_; ++d) {
        si += upd[r][d] * wi[d];
        sh += slots_s[r][d] * wh[d];
      }
      gis[o] = si;
      ghs[o] = sh;
    }
    __syncthreads();
    for (int o = tid; o < KS_ * H_; o += 256) {
      int r = o / H_, cc = o % H_;
      float ir = gis[r * 3 * H_ + cc];
      float iz = gis[r * 3 * H_ + H_ + cc];
      float in_ = gis[r * 3 * H_ + 2 * H_ + cc];
      float hr = ghs[r * 3 * H_ + cc];
      float hz = ghs[r * 3 * H_ + H_ + cc];
      float hn = ghs[r * 3 * H_ + 2 * H_ + cc];
      float rg = 1.f / (1.f + expf(-(ir + hr)));
      float zg = 1.f / (1.f + expf(-(iz + hz)));
      float ng = tanhf(in_ + rg * hn);
      slots_s[r][cc] = (1.f - zg) * ng + zg * slots_s[r][cc];
    }
    __syncthreads();

    // ---- MLP residual: h = LN(slots, g_m, b_m)
    if (wave < KS_) {
      float s1 = 0.f, s2 = 0.f;
#pragma unroll
      for (int p = 0; p < 4; ++p) {
        float x = slots_s[wave][lane + 32 * p];
        s1 += x; s2 += x * x;
      }
#pragma unroll
      for (int d = 1; d < 32; d <<= 1) {
        s1 += __shfl_xor(s1, d, 32);
        s2 += __shfl_xor(s2, d, 32);
      }
      float mu  = s1 * (1.f / 128.f);
      float var = s2 * (1.f / 128.f) - mu * mu;
      float rs  = rsqrtf(var + 1e-5f);
#pragma unroll
      for (int p = 0; p < 4; ++p) {
        int c = lane + 32 * p;
        s_ln[wave][c] = (slots_s[wave][c] - mu) * rs * g_m[c] + b_m[c];
      }
    }
    __syncthreads();
    // hidden = gelu_exact(h @ W1^T + b1)
    for (int o = tid; o < KS_ * 2 * H_; o += 256) {
      int r = o / (2 * H_), cc = o % (2 * H_);
      const float* w1r = W1 + cc * H_;
      float t = b1[cc];
#pragma unroll 4
      for (int d = 0; d < H_; ++d) t += s_ln[r][d] * w1r[d];
      hid[o] = 0.5f * t * (1.f + erff(t * 0.70710678118654752f));
    }
    __syncthreads();
    // slots += hidden @ W2^T + b2
    for (int o = tid; o < KS_ * H_; o += 256) {
      int r = o / H_, cc = o % H_;
      const float* w2r = W2 + cc * (2 * H_);
      float u = b2[cc];
#pragma unroll 4
      for (int d = 0; d < 2 * H_; ++d) u += hid[r * 2 * H_ + d] * w2r[d];
      slots_s[r][cc] += u;
    }
    __syncthreads();
  }

  for (int o = tid; o < KS_ * H_; o += 256)
    slots_out[(long)b * KS_ * H_ + o] = slots_s[o / H_][o % H_];
}

// ---------------------------------------------------------------------------
// Host launcher
// ---------------------------------------------------------------------------
extern "C" void kernel_launch(void* const* d_in, const int* in_sizes, int n_in,
                              void* d_out, int out_size, void* d_ws,
                              size_t ws_size, hipStream_t stream) {
  (void)in_sizes; (void)n_in; (void)out_size; (void)ws_size;

  const float* patch   = (const float*)d_in[0];
  const float* noise   = (const float*)d_in[1];
  const float* slot_mu = (const float*)d_in[2];
  const float* slot_ls = (const float*)d_in[3];
  const float* Wp = (const float*)d_in[4];  const float* bp  = (const float*)d_in[5];
  const float* g_in = (const float*)d_in[6]; const float* b_in = (const float*)d_in[7];
  const float* Wq = (const float*)d_in[8];  const float* bq  = (const float*)d_in[9];
  const float* Wk = (const float*)d_in[10]; const float* bk  = (const float*)d_in[11];
  const float* Wv = (const float*)d_in[12]; const float* bv  = (const float*)d_in[13];
  const float* Wih = (const float*)d_in[14]; const float* bih = (const float*)d_in[15];
  const float* Whh = (const float*)d_in[16]; const float* bhh = (const float*)d_in[17];
  const float* g_s = (const float*)d_in[18]; const float* b_s = (const float*)d_in[19];
  const float* W1 = (const float*)d_in[20]; const float* b1 = (const float*)d_in[21];
  const float* W2 = (const float*)d_in[22]; const float* b2 = (const float*)d_in[23];
  const float* g_m = (const float*)d_in[24]; const float* b_m = (const float*)d_in[25];

  // workspace layout (bytes); total ~67.6 MB (k16+v16 stay L2-resident)
  char* ws = (char*)d_ws;
  _Float16* wp16 = (_Float16*)(ws + 0);                  // 128*192*2 = 49152
  _Float16* wk16 = (_Float16*)(ws + 49152);              // 32768
  _Float16* wv16 = (_Float16*)(ws + 81920);              // 32768
  _Float16* wq16 = (_Float16*)(ws + 114688);             // 32768
  _Float16* k16  = (_Float16*)(ws + 262144);             // 128*1024*128*2
  _Float16* v16  = (_Float16*)(ws + 262144 + 33554432);  // 128*1024*128*2

  float* slots_out = (float*)d_out;                      // (B, 3, 128)
  float* attn_out  = (float*)d_out + B_ * KS_ * H_;      // (B, 3, 1024)

  cvt_w<<<96, 256, 0, stream>>>(Wp, Wk, Wv, Wq, wp16, wk16, wv16, wq16);

  proj_ln_kv<<<(B_ * NTOK_) / 128, 256, 0, stream>>>(
      patch, wp16, bp, g_in, b_in, wk16, bk, wv16, bv, k16, v16);

  slot_iter<<<B_, 256, 0, stream>>>(
      noise, slot_mu, slot_ls, wq16, bq, k16, v16,
      Wih, bih, Whh, bhh, g_s, b_s, W1, b1, W2, b2, g_m, b_m,
      slots_out, attn_out);
}
